// MultiHeadedAttention_69286412419145
// MI455X (gfx1250) — matrix operand order repaired and checked by reading, not compiled, not run on hardware
//
#include <hip/hip_runtime.h>

// =====================================================================
// Multi-head attention for gfx1250 (MI455X), bf16 WMMA + f32 accumulate
//   B=2, S=2048, D=1024, H=16, d_k=64
// =====================================================================

typedef __bf16 bf16_t;
typedef __attribute__((ext_vector_type(4)))  __bf16 v4bf;
typedef __attribute__((ext_vector_type(8)))  __bf16 v8bf;
typedef __attribute__((ext_vector_type(16))) __bf16 v16bf;
typedef __attribute__((ext_vector_type(8)))  float  v8f;

#define LDS_STRIDE 72   // 64 payload + 8 pad: conflict-free, keeps 16B alignment
#define DMODEL 1024
#define SEQ 2048

// ---------------- CDNA5 async global->LDS helpers ----------------
// Builtin signatures (from hipcc diagnostics): pointee is int vector_size(16|8),
// src in addrspace(1), dst in addrspace(3), then two immediate ints.
typedef int i4v __attribute__((vector_size(16)));
typedef int i2v __attribute__((vector_size(8)));
typedef __attribute__((address_space(1))) i4v i4v_g;
typedef __attribute__((address_space(3))) i4v i4v_l;
typedef __attribute__((address_space(1))) i2v i2v_g;
typedef __attribute__((address_space(3))) i2v i2v_l;

__device__ __forceinline__ void async_b128(const void* g, void* l) {
#if __has_builtin(__builtin_amdgcn_global_load_async_to_lds_b128)
  __builtin_amdgcn_global_load_async_to_lds_b128((i4v_g*)g, (i4v_l*)l, 0, 0);
#else
  *reinterpret_cast<v8bf*>(l) = *reinterpret_cast<const v8bf*>(g);
#endif
}

__device__ __forceinline__ void async_b64(const void* g, void* l) {
#if __has_builtin(__builtin_amdgcn_global_load_async_to_lds_b64)
  __builtin_amdgcn_global_load_async_to_lds_b64((i2v_g*)g, (i2v_l*)l, 0, 0);
#else
  *reinterpret_cast<v4bf*>(l) = *reinterpret_cast<const v4bf*>(g);
#endif
}

__device__ __forceinline__ void wait_async0() {
#if __has_builtin(__builtin_amdgcn_s_wait_asynccnt)
  __builtin_amdgcn_s_wait_asynccnt(0);
#else
  asm volatile("s_wait_asynccnt 0x0" ::: "memory");
#endif
}

// ---------------- WMMA fragment helpers ----------------
__device__ __forceinline__ v16bf cat8(v8bf lo, v8bf hi) {
  return __builtin_shufflevector(lo, hi, 0,1,2,3,4,5,6,7,8,9,10,11,12,13,14,15);
}

// A fragment: 16xK tile, row-major LDS [m][k], stride LDS_STRIDE. k0 in {0,32}.
// Per ISA: lane (m = lane&15), elems 0-7 = K[k0 + sel*8 ..], elems 8-15 = K[k0+16+sel*8 ..]
__device__ __forceinline__ v16bf load_a_frag(const bf16_t* base, int k0, int lane) {
  const int m = lane & 15, sel = lane >> 4;
  const bf16_t* p = base + m * LDS_STRIDE + k0 + sel * 8;
  v8bf lo = *reinterpret_cast<const v8bf*>(p);
  v8bf hi = *reinterpret_cast<const v8bf*>(p + 16);
  return cat8(lo, hi);
}

// B fragment: Kx16 tile stored N-major in LDS as [n][k], stride LDS_STRIDE.
// Per ISA: lane (n = lane&15), elems 0-7 = K[k0 + sel*8 ..], elems 8-15 = K[k0+16+sel*8 ..]
__device__ __forceinline__ v16bf load_b_frag(const bf16_t* base, int k0, int lane) {
  const int n = lane & 15, sel = lane >> 4;
  const bf16_t* p = base + n * LDS_STRIDE + k0 + sel * 8;
  v8bf lo = *reinterpret_cast<const v8bf*>(p);
  v8bf hi = *reinterpret_cast<const v8bf*>(p + 16);
  return cat8(lo, hi);
}

#define WMMA_BF16(a, b, c) \
  __builtin_amdgcn_wmma_f32_16x16x32_bf16(false, (a), false, (b), (short)0, (c), false, false)

// ---------------------------------------------------------------------
// GEMM: Y[m][n] = (sum_k X[m][k] * W[n][k] + bias[n]) * scale
// Block tile 128x128, 8 waves (4x2), wave tile 32x64 (2x4 WMMA accs), BK=64.
// ---------------------------------------------------------------------
template <typename TIN, typename TOUT>
__global__ __launch_bounds__(256) void gemm_xwT(const TIN* __restrict__ X,
                                                const float* __restrict__ W,
                                                const float* __restrict__ bias,
                                                TOUT* __restrict__ Y,
                                                float scale) {
  __shared__ bf16_t As[128 * LDS_STRIDE];
  __shared__ bf16_t Bs[128 * LDS_STRIDE];

  const int tid  = threadIdx.x;
  const int lane = tid & 31;
  const int wave = tid >> 5;
  const int wm   = wave & 3;   // 4 row groups of 32
  const int wn   = wave >> 2;  // 2 col groups of 64
  const int rowBase = blockIdx.y * 128;
  const int colBase = blockIdx.x * 128;

  v8f acc[2][4] = {};

  for (int kk = 0; kk < DMODEL; kk += 64) {
    // ---- stage A tile (128x64) ----
#pragma unroll
    for (int i = 0; i < 8; ++i) {
      int slot = tid + i * 256;          // 0..2047 (4 elems each)
      int r = slot >> 4, cg = slot & 15;
      if constexpr (sizeof(TIN) == 4) {
        const float4 v = *reinterpret_cast<const float4*>(
            X + (size_t)(rowBase + r) * DMODEL + kk + cg * 4);
        v4bf d;
        d[0] = (__bf16)v.x; d[1] = (__bf16)v.y; d[2] = (__bf16)v.z; d[3] = (__bf16)v.w;
        *reinterpret_cast<v4bf*>(As + r * LDS_STRIDE + cg * 4) = d;
      } else {
        // bf16 input: async DMA straight into LDS (8B chunks), no VGPR round trip
        async_b64(X + (size_t)(rowBase + r) * DMODEL + kk + cg * 4,
                  As + r * LDS_STRIDE + cg * 4);
      }
    }
    // ---- stage B tile: rows of W (output columns), naturally [n][k] ----
#pragma unroll
    for (int i = 0; i < 8; ++i) {
      int slot = tid + i * 256;
      int r = slot >> 4, cg = slot & 15;
      const float4 v = *reinterpret_cast<const float4*>(
          W + (size_t)(colBase + r) * DMODEL + kk + cg * 4);
      v4bf d;
      d[0] = (__bf16)v.x; d[1] = (__bf16)v.y; d[2] = (__bf16)v.z; d[3] = (__bf16)v.w;
      *reinterpret_cast<v4bf*>(Bs + r * LDS_STRIDE + cg * 4) = d;
    }
    if constexpr (sizeof(TIN) == 2) wait_async0();
    __syncthreads();

#pragma unroll
    for (int ks = 0; ks < 64; ks += 32) {
      v16bf a[2], b[4];
#pragma unroll
      for (int tm = 0; tm < 2; ++tm)
        a[tm] = load_a_frag(As + (wm * 32 + tm * 16) * LDS_STRIDE, ks, lane);
#pragma unroll
      for (int tn = 0; tn < 4; ++tn)
        b[tn] = load_b_frag(Bs + (wn * 64 + tn * 16) * LDS_STRIDE, ks, lane);
#pragma unroll
      for (int tm = 0; tm < 2; ++tm)
#pragma unroll
        for (int tn = 0; tn < 4; ++tn)
          acc[tm][tn] = WMMA_BF16(a[tm], b[tn], acc[tm][tn]);
    }
    __syncthreads();
  }

  // ---- epilogue: bias + scale + store ----
  float bv[4];
#pragma unroll
  for (int tn = 0; tn < 4; ++tn)
    bv[tn] = bias[colBase + wn * 64 + tn * 16 + (lane & 15)];

#pragma unroll
  for (int tm = 0; tm < 2; ++tm) {
#pragma unroll
    for (int tn = 0; tn < 4; ++tn) {
#pragma unroll
      for (int e = 0; e < 8; ++e) {
        int gr = rowBase + wm * 32 + tm * 16 + e + 8 * (lane >> 4);
        int gc = colBase + wn * 64 + tn * 16 + (lane & 15);
        float v = (acc[tm][tn][e] + bv[tn]) * scale;
        if constexpr (sizeof(TOUT) == 4)
          Y[(size_t)gr * DMODEL + gc] = v;
        else
          Y[(size_t)gr * DMODEL + gc] = (bf16_t)v;
      }
    }
  }
}

// ---------------------------------------------------------------------
// Flash attention per (batch, head). Grid: (S/128, H, B), 256 threads.
// Each wave owns 16 query rows; key blocks of 64 with online softmax.
// Q is pre-scaled by 1/sqrt(d_k)*log2(e), so WMMA scores are exp2-ready.
// ---------------------------------------------------------------------
__global__ __launch_bounds__(256) void attn(const bf16_t* __restrict__ Qp,
                                            const bf16_t* __restrict__ Kp,
                                            const bf16_t* __restrict__ Vp,
                                            bf16_t* __restrict__ Xo) {
  __shared__ bf16_t Qs[128 * LDS_STRIDE];     // 18 KB
  __shared__ bf16_t Ks[64 * LDS_STRIDE];      //  9 KB
  __shared__ bf16_t Vts[64 * LDS_STRIDE];     //  9 KB  (V^T: [d][s])
  __shared__ bf16_t Ps[8][16 * LDS_STRIDE];   // 18 KB  per-wave P tiles

  const int tid  = threadIdx.x;
  const int lane = tid & 31;
  const int wave = tid >> 5;
  const int qb = blockIdx.x, h = blockIdx.y, b = blockIdx.z;

  const bf16_t* Qg = Qp + ((size_t)(b * SEQ + qb * 128)) * DMODEL + h * 64;
  const bf16_t* Kg = Kp + ((size_t)(b * SEQ)) * DMODEL + h * 64;
  const bf16_t* Vg = Vp + ((size_t)(b * SEQ)) * DMODEL + h * 64;

  // async-load Q block (128x64 bf16) into LDS, 16B chunks (per-lane LDS addrs)
#pragma unroll
  for (int i = 0; i < 4; ++i) {
    int slot = tid + i * 256;  // 0..1023
    int r = slot >> 3, cg = slot & 7;
    async_b128(Qg + (size_t)r * DMODEL + cg * 8, Qs + r * LDS_STRIDE + cg * 8);
  }

  float Mrun[8], Lrun[8];
#pragma unroll
  for (int e = 0; e < 8; ++e) { Mrun[e] = -1e30f; Lrun[e] = 0.0f; }
  v8f o[4] = {};

  for (int kb = 0; kb < SEQ; kb += 64) {
    // async-stage K block (64x64) row-major
#pragma unroll
    for (int i = 0; i < 2; ++i) {
      int slot = tid + i * 256;  // 0..511
      int r = slot >> 3, cg = slot & 7;
      async_b128(Kg + (size_t)(kb + r) * DMODEL + cg * 8,
                 Ks + r * LDS_STRIDE + cg * 8);
    }
    // stage V block transposed -> Vts[d][s] (VGPR path: needs transpose)
#pragma unroll
    for (int i = 0; i < 2; ++i) {
      int slot = tid + i * 256;
      int r = slot >> 3, cg = slot & 7;
      v8bf v = *reinterpret_cast<const v8bf*>(Vg + (size_t)(kb + r) * DMODEL + cg * 8);
#pragma unroll
      for (int j = 0; j < 8; ++j) Vts[(cg * 8 + j) * LDS_STRIDE + r] = v[j];
    }
    wait_async0();
    __syncthreads();

    // ---- scores (already in exp2 domain): S = Q_tile x K_block^T ----
    v8f sc[4] = {};
#pragma unroll
    for (int ks = 0; ks < 64; ks += 32) {
      v16bf a = load_a_frag(Qs + (wave * 16) * LDS_STRIDE, ks, lane);
#pragma unroll
      for (int tc = 0; tc < 4; ++tc) {
        v16bf bf = load_b_frag(Ks + (tc * 16) * LDS_STRIDE, ks, lane);
        sc[tc] = WMMA_BF16(a, bf, sc[tc]);
      }
    }

    // ---- online softmax: row r = e + 8*(lane>>4), col = (lane&15)+16*tc ----
    float mloc[8];
#pragma unroll
    for (int e = 0; e < 8; ++e)
      mloc[e] = fmaxf(fmaxf(sc[0][e], sc[1][e]), fmaxf(sc[2][e], sc[3][e]));
#pragma unroll
    for (int msk = 1; msk <= 8; msk <<= 1)
#pragma unroll
      for (int e = 0; e < 8; ++e)
        mloc[e] = fmaxf(mloc[e], __shfl_xor(mloc[e], msk, 32));

    float alpha[8], rs[8];
#pragma unroll
    for (int e = 0; e < 8; ++e) {
      float Mn = fmaxf(Mrun[e], mloc[e]);
      alpha[e] = __builtin_amdgcn_exp2f(Mrun[e] - Mn);
      Mrun[e] = Mn;
      rs[e] = 0.0f;
    }

    bf16_t* Pw = Ps[wave];
#pragma unroll
    for (int tc = 0; tc < 4; ++tc)
#pragma unroll
      for (int e = 0; e < 8; ++e) {
        float p = __builtin_amdgcn_exp2f(sc[tc][e] - Mrun[e]);
        rs[e] += p;
        Pw[(e + 8 * (lane >> 4)) * LDS_STRIDE + tc * 16 + (lane & 15)] = (bf16_t)p;
      }
#pragma unroll
    for (int msk = 1; msk <= 8; msk <<= 1)
#pragma unroll
      for (int e = 0; e < 8; ++e) rs[e] += __shfl_xor(rs[e], msk, 32);
#pragma unroll
    for (int e = 0; e < 8; ++e) Lrun[e] = Lrun[e] * alpha[e] + rs[e];
#pragma unroll
    for (int dt = 0; dt < 4; ++dt)
#pragma unroll
      for (int e = 0; e < 8; ++e) o[dt][e] *= alpha[e];

    // ---- O += P (16x64) x V_block (64x64)  (B from Vts = [d][s]) ----
#pragma unroll
    for (int ks = 0; ks < 64; ks += 32) {
      v16bf a = load_a_frag(Pw, ks, lane);
#pragma unroll
      for (int dt = 0; dt < 4; ++dt) {
        v16bf bf = load_b_frag(Vts + (dt * 16) * LDS_STRIDE, ks, lane);
        o[dt] = WMMA_BF16(a, bf, o[dt]);
      }
    }
    __syncthreads();
  }

  // ---- normalize and write merged-head output (bf16 [B*S][D]) ----
  float inv[8];
#pragma unroll
  for (int e = 0; e < 8; ++e) inv[e] = 1.0f / Lrun[e];
#pragma unroll
  for (int dt = 0; dt < 4; ++dt)
#pragma unroll
    for (int e = 0; e < 8; ++e) {
      int r = e + 8 * (lane >> 4);
      size_t row = (size_t)(b * SEQ + qb * 128 + wave * 16 + r);
      Xo[row * DMODEL + h * 64 + dt * 16 + (lane & 15)] = (bf16_t)(o[dt][e] * inv[e]);
    }
}

// ---------------------------------------------------------------------
extern "C" void kernel_launch(void* const* d_in, const int* in_sizes, int n_in,
                              void* d_out, int out_size, void* d_ws, size_t ws_size,
                              hipStream_t stream) {
  (void)in_sizes; (void)n_in; (void)out_size; (void)ws_size;
  const float* q = (const float*)d_in[0];
  const float* k = (const float*)d_in[1];
  const float* v = (const float*)d_in[2];
  const float* W = (const float*)d_in[3];
  const float* bias = (const float*)d_in[4];
  float* out = (float*)d_out;

  const size_t NTOK = (size_t)2 * SEQ;          // 4096 rows
  bf16_t* Qp = (bf16_t*)d_ws;
  bf16_t* Kp = Qp + NTOK * DMODEL;
  bf16_t* Vp = Kp + NTOK * DMODEL;
  bf16_t* Xa = Vp + NTOK * DMODEL;

  dim3 gGemm(DMODEL / 128, NTOK / 128);          // (8, 32)
  dim3 gAttn(SEQ / 128, 16, 2);                  // (16, 16, 2)

  const float SCL = 0.125f * 1.4426950408889634f;  // 1/sqrt(64) * log2(e)

  gemm_xwT<float, bf16_t><<<gGemm, 256, 0, stream>>>(q, W, bias, Qp, SCL);
  gemm_xwT<float, bf16_t><<<gGemm, 256, 0, stream>>>(k, W, bias, Kp, 1.0f);
  gemm_xwT<float, bf16_t><<<gGemm, 256, 0, stream>>>(v, W, bias, Vp, 1.0f);
  attn<<<gAttn, 256, 0, stream>>>(Qp, Kp, Vp, Xa);
  gemm_xwT<bf16_t, float><<<gGemm, 256, 0, stream>>>(Xa, W, bias, out, 1.0f);
}
